// RoPEMultiHeadAttention_79551384256916
// MI455X (gfx1250) — matrix operand order repaired and checked by reading, not compiled, not run on hardware
//
#include <hip/hip_runtime.h>

// ---------------------------------------------------------------------------
// RoPE Multi-Head Attention forward for gfx1250 (MI455X, wave32, WMMA, TDM)
//   S=2048, B=2, E=1024, H=16, D=64
// Phase 0: cvt_f16_kernel     -> f16 copies of X (q,k,v inputs) and W matrices
// Phase 1: proj_rope_kernel   -> q,k (rope'd, [b,h,s,d] f16), v ([b,h,d,s] f16)
// Phase 2: flash_attn_kernel  -> TDM-staged K/V, online softmax, fp32 output
// ---------------------------------------------------------------------------

typedef __attribute__((ext_vector_type(16))) _Float16      v16h;
typedef __attribute__((ext_vector_type(8)))  _Float16      v8h;
typedef __attribute__((ext_vector_type(8)))  float         v8f;
typedef __attribute__((ext_vector_type(4)))  unsigned int  v4u;
typedef __attribute__((ext_vector_type(4)))  int           v4i;
typedef __attribute__((ext_vector_type(8)))  int           v8i;

#define DEV __device__ __forceinline__

// load 8 consecutive f16 (16 bytes) at half-positions [POS..POS+7]
template<int POS>
DEV void load8h(v16h& v, const _Float16* p) {
    v8h t = *(const v8h*)(p);
#pragma unroll
    for (int i = 0; i < 8; ++i) v[POS+i] = t[i];
}

DEV v8f wmma_f16(const v16h& a, const v16h& b, const v8f& c) {
    return __builtin_amdgcn_wmma_f32_16x16x32_f16(false, a, false, b,
                                                  (short)0, c, false, false);
}

// ---------------------------------------------------------------------------
// TDM 2D tile load: global (row-major, 2-byte elems, stride0 elems per row)
// into LDS at lds_addr. D# packing per CDNA5 ISA.
// ---------------------------------------------------------------------------
DEV void tdm_load_2d(unsigned int lds_addr, const void* gptr,
                     unsigned int tensor_d0, unsigned int tensor_d1,
                     unsigned int tile_d0,   unsigned int tile_d1,
                     unsigned long long stride0)
{
    const unsigned long long ga = (unsigned long long)(uintptr_t)gptr;
    v4u g0;
    g0[0] = 1u;                                            // count=1 (valid), user mode
    g0[1] = lds_addr;                                      // LDS byte address
    g0[2] = (unsigned int)(ga & 0xffffffffu);              // global_addr[31:0]
    g0[3] = (unsigned int)((ga >> 32) & 0x01ffffffu) | (2u << 30);  // addr[56:32] | type=2
    v8i g1;
    g1[0] = (int)(1u << 16);                               // data_size=1 (2B), wg_mask=0
    g1[1] = (int)((tensor_d0 & 0xffffu) << 16);            // tensor_dim0[15:0]
    g1[2] = (int)((tensor_d0 >> 16) | ((tensor_d1 & 0xffffu) << 16));
    g1[3] = (int)((tensor_d1 >> 16) | (tile_d0 << 16));    // tile_dim0
    g1[4] = (int)(tile_d1 & 0xffffu);                      // tile_dim1 (tile_dim2=0)
    g1[5] = (int)(unsigned int)(stride0 & 0xffffffffu);    // tensor_dim0_stride[31:0]
    g1[6] = (int)(unsigned int)((stride0 >> 32) & 0xffffu);// stride[47:32]; dim1_stride=0
    g1[7] = 0;
    v4i z4 = {0, 0, 0, 0};
#if __clang_major__ >= 23
    v8i z8 = {0, 0, 0, 0, 0, 0, 0, 0};
    __builtin_amdgcn_tensor_load_to_lds(g0, g1, z4, z4, z8, 0);
#else
    __builtin_amdgcn_tensor_load_to_lds(g0, g1, z4, z4, 0);
#endif
}

// ---------------------------------------------------------------------------
// Phase 0: one-shot fp32 -> f16 convert (removes per-K-step cvt VALU from GEMM
// and halves the W/X bytes re-read from L2/HBM by the projection kernel).
// ---------------------------------------------------------------------------
__global__ __launch_bounds__(256)
void cvt_f16_kernel(const float* __restrict__ src, _Float16* __restrict__ dst)
{
    const int i = (blockIdx.x * 256 + threadIdx.x) * 8;
    float4 a = *(const float4*)(src + i);
    float4 b = *(const float4*)(src + i + 4);
    v8h o;
    o[0]=(_Float16)a.x; o[1]=(_Float16)a.y; o[2]=(_Float16)a.z; o[3]=(_Float16)a.w;
    o[4]=(_Float16)b.x; o[5]=(_Float16)b.y; o[6]=(_Float16)b.z; o[7]=(_Float16)b.w;
    *(v8h*)(dst + i) = o;
}

// ---------------------------------------------------------------------------
// Phase 1: Y = X @ W^T + bias, fused RoPE (mode 0/1) or transposed store (mode 2)
// One wave computes a 16(tokens) x 64(one head) strip: 4 accumulators.
// Grid: 4096 waves = 256 token-tiles x 16 heads; 1024 blocks x 128 threads.
// ---------------------------------------------------------------------------
__global__ __launch_bounds__(128)
void proj_rope_kernel(const _Float16* __restrict__ X, const _Float16* __restrict__ W,
                      const float* __restrict__ bias, _Float16* __restrict__ dst,
                      int mode)
{
    const int lane = threadIdx.x & 31;
    const int lm = lane & 15, lh = lane >> 4;
    const int wave  = blockIdx.x * (blockDim.x >> 5) + (threadIdx.x >> 5);
    const int mtile = wave & 255;   // 256 tiles of 16 tokens (4096 tokens)
    const int strip = wave >> 8;    // head index 0..15
    const int t0 = mtile * 16;
    const int e0 = strip * 64;

    v8f acc[4] = {};

    for (int kb = 0; kb < 1024; kb += 32) {
        // A (16x32 f16): lane(lm)=row m; k = lh*16 + 0..15 contiguous
        v16h A;
        const _Float16* ap = X + (size_t)(t0 + lm) * 1024 + kb + lh * 16;
        __builtin_prefetch(ap + 32, 0, 1);           // next K-step
        load8h<0>(A, ap);
        load8h<8>(A, ap + 8);
#pragma unroll
        for (int sub = 0; sub < 4; ++sub) {
            // B (32x16 f16): lane(lm)=col n -> W row e; k = lh*16 + 0..15 contiguous
            const _Float16* bp = W + (size_t)(e0 + sub*16 + lm) * 1024 + kb + lh * 16;
            v16h Bt;
            load8h<0>(Bt, bp);
            load8h<8>(Bt, bp + 8);
            acc[sub] = wmma_f16(A, Bt, acc[sub]);
        }
    }

    float bv[4];
#pragma unroll
    for (int sub = 0; sub < 4; ++sub) bv[sub] = bias[e0 + sub*16 + lm];

    if (mode == 2) {
        // V: store transposed [b, h, d, s] (feeds P@V B-operand contiguously)
#pragma unroll
        for (int sub = 0; sub < 4; ++sub) {
            const int d = sub*16 + lm;
#pragma unroll
            for (int r = 0; r < 8; ++r) {
                const int t = t0 + r + lh*8;          // token = s*2 + b
                const int b = t & 1, s = t >> 1;
                const float val = acc[sub][r] + bv[sub];
                dst[((size_t)(b*16 + strip) * 64 + d) * 2048 + s] = (_Float16)val;
            }
        }
    } else {
        // Q/K: fused RoPE. Pair (d, d+-32) lives in acc[sub^2], same lane/VGPR.
        const float NEG_LN_BASE_OVER_32 = -0.2878231366242557f;  // -ln(10000)/32
#pragma unroll
        for (int sub = 0; sub < 4; ++sub) {
            const int d  = sub*16 + lm;
            const int dm = d & 31;
            const float freq = __expf((float)dm * NEG_LN_BASE_OVER_32);
            const int psub = sub ^ 2;
#pragma unroll
            for (int r = 0; r < 8; ++r) {
                const int t = t0 + r + lh*8;
                const int b = t & 1, s = t >> 1;
                float sn, cs;
                __sincosf((float)s * freq, &sn, &cs);
                const float x   = acc[sub][r]  + bv[sub];
                const float xp  = acc[psub][r] + bv[psub];
                const float rot = (sub < 2) ? -xp : xp;  // rotate_half
                const float val = x * cs + rot * sn;
                dst[((size_t)(b*16 + strip) * 2048 + s) * 64 + d] = (_Float16)val;
            }
        }
    }
}

// ---------------------------------------------------------------------------
// Phase 2: flash attention. All 4 waves of a block share one (b,h); K/V blocks
// (32 keys) are TDM-staged into LDS once per step and shared (4x L2 traffic cut).
// Grid: 4096 waves = 32 (b,h) x 128 q-tiles; 1024 blocks x 128 threads.
// ---------------------------------------------------------------------------
__global__ __launch_bounds__(128)
void flash_attn_kernel(const _Float16* __restrict__ qh,
                       const _Float16* __restrict__ kh,
                       const _Float16* __restrict__ vt,
                       float* __restrict__ out)
{
    __shared__ __align__(16) _Float16 kbuf[32 * 64];     // [key][d]   4 KB
    __shared__ __align__(16) _Float16 vbuf[64 * 32];     // [d][key]   4 KB
    __shared__ __align__(16) _Float16 pbuf[4][16 * 32];  // per-wave P staging

    const int lane = threadIdx.x & 31;
    const int lm = lane & 15, lh = lane >> 4;
    const int wv = threadIdx.x >> 5;
    const int gw = blockIdx.x * 4 + wv;
    const int bh = gw >> 7;     // 0..31 (b*16 + h); uniform across the block
    const int qt = gw & 127;
    const int s0 = qt * 16;

    // Q (16x64) in A layout: two 16x32 operands, loaded once
    v16h A0, A1;
    {
        const _Float16* qp = qh + ((size_t)bh * 2048 + s0 + lm) * 64 + lh * 16;
        load8h<0>(A0, qp);      load8h<8>(A0, qp + 8);
        load8h<0>(A1, qp + 32); load8h<8>(A1, qp + 40);
    }

    v8f O[4] = {};
    float rm[8], rl[8];
#pragma unroll
    for (int r = 0; r < 8; ++r) { rm[r] = -1e30f; rl[r] = 0.f; }

    _Float16* lp = &pbuf[wv][0];
    const unsigned int klds = (unsigned int)(uintptr_t)&kbuf[0];
    const unsigned int vlds = (unsigned int)(uintptr_t)&vbuf[0];
    const _Float16* kgbase = kh + (size_t)bh * 2048 * 64;
    const _Float16* vgbase = vt + (size_t)bh * 64 * 2048;

    for (int j = 0; j < 2048; j += 32) {
        // ---- TDM: stage K block (32x64, contiguous) and V block (64 rows x 32,
        //      stride 2048) into LDS; one wave issues, block barriers. ----
        if (wv == 0) {
            tdm_load_2d(klds, kgbase + (size_t)j * 64,
                        /*tensor_d0=*/2048 * 64, /*tensor_d1=*/1,
                        /*tile_d0=*/2048, /*tile_d1=*/1, /*stride0=*/2048ull * 64);
            tdm_load_2d(vlds, vgbase + j,
                        /*tensor_d0=*/2048, /*tensor_d1=*/64,
                        /*tile_d0=*/32, /*tile_d1=*/64, /*stride0=*/2048ull);
            __builtin_amdgcn_s_wait_tensorcnt(0);
        }
        __syncthreads();

        // ---- scores: S(16x32) = Q(16x64) . K(32 keys x64)^T ----
        v16h B0, B1;
        v8f Sa = {}, Sb = {};
        {
            const _Float16* kp = &kbuf[lm * 64 + lh * 16];
            load8h<0>(B0, kp);      load8h<8>(B0, kp + 8);
            load8h<0>(B1, kp + 32); load8h<8>(B1, kp + 40);
            Sa = wmma_f16(A0, B0, Sa);
            Sa = wmma_f16(A1, B1, Sa);
            const _Float16* kp2 = kp + 16 * 64;
            load8h<0>(B0, kp2);      load8h<8>(B0, kp2 + 8);
            load8h<0>(B1, kp2 + 32); load8h<8>(B1, kp2 + 40);
            Sb = wmma_f16(A0, B0, Sb);
            Sb = wmma_f16(A1, B1, Sb);
        }

        // ---- online softmax (row = r + lh*8; N-group = lanes sharing lh) ----
#pragma unroll
        for (int r = 0; r < 8; ++r) {
            const float sa = Sa[r] * 0.125f;   // 1/sqrt(64)
            const float sb = Sb[r] * 0.125f;
            float mx = fmaxf(sa, sb);
            mx = fmaxf(mx, __shfl_xor(mx, 1, 32));
            mx = fmaxf(mx, __shfl_xor(mx, 2, 32));
            mx = fmaxf(mx, __shfl_xor(mx, 4, 32));
            mx = fmaxf(mx, __shfl_xor(mx, 8, 32));
            const float mnew  = fmaxf(rm[r], mx);
            const float alpha = __expf(rm[r] - mnew);
            rm[r] = mnew;
            const float pa = __expf(sa - mnew);
            const float pb = __expf(sb - mnew);
            float rs = pa + pb;
            rs += __shfl_xor(rs, 1, 32);
            rs += __shfl_xor(rs, 2, 32);
            rs += __shfl_xor(rs, 4, 32);
            rs += __shfl_xor(rs, 8, 32);
            rl[r] = rl[r] * alpha + rs;
            O[0][r] *= alpha; O[1][r] *= alpha; O[2][r] *= alpha; O[3][r] *= alpha;
            const int row = r + lh * 8;
            lp[row * 32 + lm]      = (_Float16)pa;   // P row-major [16][32]
            lp[row * 32 + 16 + lm] = (_Float16)pb;
        }
        asm volatile("s_wait_dscnt 0" ::: "memory");  // same-wave DS in-order + fence

        // ---- P back in A layout (16x32) ----
        v16h P;
        {
            const _Float16* pp = lp + lm * 32 + lh * 16;
            load8h<0>(P, pp);
            load8h<8>(P, pp + 8);
        }

        // ---- O(16x64) += P(16x32) . V(32x64) from LDS [d][key] ----
#pragma unroll
        for (int sub = 0; sub < 4; ++sub) {
            const _Float16* vp = &vbuf[(sub * 16 + lm) * 32 + lh * 16];
            v16h Bv;
            load8h<0>(Bv, vp);
            load8h<8>(Bv, vp + 8);
            O[sub] = wmma_f16(P, Bv, O[sub]);
        }
        __syncthreads();   // before next step's TDM overwrites kbuf/vbuf
    }

    // ---- epilogue: normalize, scatter to [S, B, 1024] fp32 ----
    const int b = bh >> 4, h = bh & 15;
#pragma unroll
    for (int sub = 0; sub < 4; ++sub) {
        const int d = sub*16 + lm;
#pragma unroll
        for (int r = 0; r < 8; ++r) {
            const int s = s0 + r + lh * 8;
            out[((size_t)s * 2 + b) * 1024 + h * 64 + d] = O[sub][r] / rl[r];
        }
    }
}

// ---------------------------------------------------------------------------
extern "C" void kernel_launch(void* const* d_in, const int* in_sizes, int n_in,
                              void* d_out, int out_size, void* d_ws, size_t ws_size,
                              hipStream_t stream) {
    (void)in_sizes; (void)n_in; (void)out_size; (void)ws_size;
    const float* q  = (const float*)d_in[0];
    const float* k  = (const float*)d_in[1];
    const float* v  = (const float*)d_in[2];
    const float* Wq = (const float*)d_in[3];
    const float* bq = (const float*)d_in[4];
    const float* Wk = (const float*)d_in[5];
    const float* bk = (const float*)d_in[6];
    const float* Wv = (const float*)d_in[7];
    const float* bv = (const float*)d_in[8];

    const size_t NX = (size_t)4096 * 1024;   // tokens x d_model
    const size_t NW = (size_t)1024 * 1024;
    _Float16* qh = (_Float16*)d_ws;          // [b,h,s,d] rope'd
    _Float16* kh = qh + NX;
    _Float16* vt = kh + NX;                  // [b,h,d,s]
    _Float16* xq = vt + NX;                  // f16 copies of inputs
    _Float16* xk = xq + NX;
    _Float16* xv = xk + NX;
    _Float16* wq = xv + NX;                  // f16 copies of weights
    _Float16* wk = wq + NW;
    _Float16* wv = wk + NW;

    dim3 blk(128);   // 4 waves (wave32)

    // Phase 0: f32 -> f16 (8 elems/thread)
    cvt_f16_kernel<<<NX / 8 / 256, 256, 0, stream>>>(q,  xq);
    cvt_f16_kernel<<<NX / 8 / 256, 256, 0, stream>>>(k,  xk);
    cvt_f16_kernel<<<NX / 8 / 256, 256, 0, stream>>>(v,  xv);
    cvt_f16_kernel<<<NW / 8 / 256, 256, 0, stream>>>(Wq, wq);
    cvt_f16_kernel<<<NW / 8 / 256, 256, 0, stream>>>(Wk, wk);
    cvt_f16_kernel<<<NW / 8 / 256, 256, 0, stream>>>(Wv, wv);

    // Phase 1: projections (+bias, +RoPE for q/k, transposed store for v)
    proj_rope_kernel<<<1024, blk, 0, stream>>>(xq, wq, bq, qh, 0);
    proj_rope_kernel<<<1024, blk, 0, stream>>>(xk, wk, bk, kh, 1);
    proj_rope_kernel<<<1024, blk, 0, stream>>>(xv, wv, bv, vt, 2);

    // Phase 2: attention
    flash_attn_kernel<<<1024, blk, 0, stream>>>(qh, kh, vt, (float*)d_out);
}
